// CopyGenerator_49692771614991
// MI455X (gfx1250) — matrix operand order repaired and checked by reading, not compiled, not run on hardware
//
#include <hip/hip_runtime.h>
#include <hip/hip_bf16.h>
#include <stdint.h>

#define B_   8
#define TQ_  64
#define TK_  128
#define D_   512
#define V_   32000
#define EXT_ (V_ + TK_)
#define MROWS (B_ * TQ_)   /* 512  */
#define KROWS (B_ * TK_)   /* 1024 */

typedef __attribute__((ext_vector_type(16))) __bf16 v16bf;
typedef __attribute__((ext_vector_type(8)))  float  v8f;

union Frag { v16bf v; uint4 q[2]; };

__device__ __forceinline__ unsigned short f2bf(float f) {
  unsigned int u = __float_as_uint(f);
  u += 0x7FFFu + ((u >> 16) & 1u);          // round-to-nearest-even
  return (unsigned short)(u >> 16);
}

// CDNA5 async global->LDS copy (16B per lane), tracked by ASYNCcnt.
__device__ __forceinline__ void async_copy16(const unsigned short* g,
                                             const unsigned short* l) {
  asm volatile("global_load_async_to_lds_b128 %0, %1, off"
               :: "v"((unsigned int)(uintptr_t)l), "v"(g)
               : "memory");
}
__device__ __forceinline__ void wait_async0() {
  asm volatile("s_wait_asynccnt 0x0" ::: "memory");
}

// ---------------------------------------------------------------- convert
__global__ void k_convert(const float* __restrict__ src,
                          unsigned short* __restrict__ dst, int n) {
  int i = blockIdx.x * blockDim.x + threadIdx.x;
  int stride = gridDim.x * blockDim.x;
  for (; i < n; i += stride) dst[i] = f2bf(src[i]);
}

// ---------------------------------------------------------------- masks
__global__ void k_masks(const float* __restrict__ tgt, const float* __restrict__ src,
                        float* __restrict__ tmask, float* __restrict__ smask) {
  __shared__ float red[256];
  const int row = blockIdx.x, t = threadIdx.x;
  const float* p = (row < MROWS) ? (tgt + (size_t)row * D_)
                                 : (src + (size_t)(row - MROWS) * D_);
  red[t] = fabsf(p[t]) + fabsf(p[t + 256]);
  __syncthreads();
  for (int o = 128; o > 0; o >>= 1) { if (t < o) red[t] += red[t + o]; __syncthreads(); }
  if (t == 0) {
    float m = (red[0] > 0.f) ? 1.f : 0.f;
    if (row < MROWS) tmask[row] = m; else smask[row - MROWS] = m;
  }
}

// ---------------------------------------------------------------- bf16 WMMA GEMM
// C[m][n] = sum_k A[m][k] * Bm[n][k]  (+ bias[n]),  tile 64(M) x 256(N) x 64(K),
// double-buffered LDS filled with GLOBAL_LOAD_ASYNC_TO_LDS_B128.
__global__ __launch_bounds__(256) void k_gemm_bf16(
    const unsigned short* __restrict__ A, int lda,
    const unsigned short* __restrict__ Bm, int ldb,
    float* __restrict__ C, long long ldc,
    int K, const float* __restrict__ bias) {
  __shared__ __align__(16) unsigned short As[2][64 * 64];
  __shared__ __align__(16) unsigned short Bs[2][256 * 64];
  const int tid  = threadIdx.x;
  const int wave = tid >> 5, lane = tid & 31;
  const int h = lane >> 4, l16 = lane & 15;
  const int mBase = blockIdx.y * 64;
  const int nBase = blockIdx.x * 256;
  v8f acc[4][2] = {};

  auto load_tiles = [&](int k0, int buf) {
    for (int c = tid; c < 512; c += 256) {              // A tile: 64x64 bf16
      int row = c >> 3, col = (c & 7) << 3;
      async_copy16(&A[(size_t)(mBase + row) * lda + k0 + col],
                   &As[buf][row * 64 + col]);
    }
    for (int c = tid; c < 2048; c += 256) {             // B tile: 256x64 bf16
      int row = c >> 3, col = (c & 7) << 3;
      async_copy16(&Bm[(size_t)(nBase + row) * ldb + k0 + col],
                   &Bs[buf][row * 64 + col]);
    }
  };

  int buf = 0;
  load_tiles(0, 0);                                     // prologue fill

  for (int k0 = 0; k0 < K; k0 += 64) {
    wait_async0();                                      // own wave's fills landed
    __syncthreads();                                    // everyone's fills landed;
                                                        // prev buffer fully consumed
    if (k0 + 64 < K) load_tiles(k0 + 64, buf ^ 1);      // overlap next slab with WMMAs

    for (int s = 0; s < 2; ++s) {
      const int kk = s << 5;
      Frag a[4], b[2];
#pragma unroll
      for (int i = 0; i < 4; ++i) {                     // A frag: K = 8h..8h+7, 16+8h..
        const unsigned short* pa = &As[buf][(i * 16 + l16) * 64 + kk + h * 8];
        a[i].q[0] = *(const uint4*)pa;
        a[i].q[1] = *(const uint4*)(pa + 16);
      }
#pragma unroll
      for (int j = 0; j < 2; ++j) {                     // B frag: K = 8h..8h+7, 16+8h..
        const unsigned short* pb = &Bs[buf][(wave * 32 + j * 16 + l16) * 64 + kk + h * 8];
        b[j].q[0] = *(const uint4*)pb;
        b[j].q[1] = *(const uint4*)(pb + 16);
      }
#pragma unroll
      for (int i = 0; i < 4; ++i)
#pragma unroll
        for (int j = 0; j < 2; ++j)
          acc[i][j] = __builtin_amdgcn_wmma_f32_16x16x32_bf16(
              false, a[i].v, false, b[j].v, (short)0, acc[i][j], false, false);
    }
    buf ^= 1;
  }

#pragma unroll
  for (int j = 0; j < 2; ++j) {
    const int n = nBase + wave * 32 + j * 16 + l16;
    const float bv = bias ? bias[n] : 0.f;
#pragma unroll
    for (int i = 0; i < 4; ++i)
#pragma unroll
      for (int r = 0; r < 8; ++r)                       // C: M = r + 8h, N = l16
        C[(size_t)(mBase + i * 16 + h * 8 + r) * ldc + n] = acc[i][j][r] + bv;
  }
}

// ---------------------------------------------------------------- per-row mean/rstd over V
__global__ void k_rowstats(const float* __restrict__ C, float* __restrict__ stats) {
  __shared__ float rs[256], rq[256];
  const int row = blockIdx.x, t = threadIdx.x;
  const float* p = C + (size_t)row * EXT_;
  float s = 0.f, q = 0.f;
  for (int c = t; c < V_; c += 256) { float x = p[c]; s += x; q += x * x; }
  rs[t] = s; rq[t] = q; __syncthreads();
  for (int o = 128; o > 0; o >>= 1) {
    if (t < o) { rs[t] += rs[t + o]; rq[t] += rq[t + o]; }
    __syncthreads();
  }
  if (t == 0) {
    float mean = rs[0] / (float)V_;
    float var  = rq[0] / (float)V_ - mean * mean;
    stats[row * 2]     = mean;
    stats[row * 2 + 1] = rsqrtf(var + 1e-5f);
  }
}

// ---------------------------------------------------------------- additive attention
__global__ __launch_bounds__(256) void k_attention(
    const float* __restrict__ qproj, const float* __restrict__ kproj,
    const float* __restrict__ src_key,
    const float* __restrict__ b_attn, const float* __restrict__ v_w,
    const float* __restrict__ W_lin, const float* __restrict__ b_lin,
    const float* __restrict__ tmask, const float* __restrict__ smask,
    float* __restrict__ att_out, float* __restrict__ p_out) {
  __shared__ float qs[512], wgt[512], att[128], sm[128], red[256];
  const int bq = blockIdx.x, t = threadIdx.x;
  const int b = bq >> 6;
  const float tgtm = tmask[bq];
  qs[t]       = qproj[(size_t)bq * D_ + t]       + b_attn[t];
  qs[t + 256] = qproj[(size_t)bq * D_ + t + 256] + b_attn[t + 256];
  __syncthreads();

  const int wave = t >> 5, lane = t & 31;
  for (int kk = wave; kk < TK_; kk += 8) {              // tanh-attention scores
    const float* kp = kproj + (size_t)(b * TK_ + kk) * D_;
    float s = 0.f;
    for (int d = lane; d < D_; d += 32) s += tanhf(qs[d] + kp[d]) * v_w[d];
    for (int o = 16; o > 0; o >>= 1) s += __shfl_xor(s, o, 32);
    if (lane == 0) att[kk] = s;
  }
  __syncthreads();

  float aval = 0.f, smK = 0.f, amask = -__builtin_inff();
  if (t < TK_) {
    smK = smask[b * TK_ + t];
    aval = att[t];
    amask = (smK == 0.f) ? -__builtin_inff() : aval;
  }
  red[t] = amask; __syncthreads();                      // softmax max
  for (int o = 128; o > 0; o >>= 1) { if (t < o) red[t] = fmaxf(red[t], red[t + o]); __syncthreads(); }
  const float mx = red[0]; __syncthreads();
  const float ev = (t < TK_) ? __expf(amask - mx) : 0.f;
  red[t] = ev; __syncthreads();                         // softmax sum
  for (int o = 128; o > 0; o >>= 1) { if (t < o) red[t] += red[t + o]; __syncthreads(); }
  const float esum = red[0]; __syncthreads();
  if (t < TK_) sm[t] = ev / esum;

  const float oav = (t < TK_) ? aval * smK * tgtm : 0.f; // out_att + LN over TK
  red[t] = oav; __syncthreads();
  for (int o = 128; o > 0; o >>= 1) { if (t < o) red[t] += red[t + o]; __syncthreads(); }
  const float omean = red[0] / (float)TK_; __syncthreads();
  const float dv = (t < TK_) ? (oav - omean) : 0.f;
  red[t] = dv * dv; __syncthreads();
  for (int o = 128; o > 0; o >>= 1) { if (t < o) red[t] += red[t + o]; __syncthreads(); }
  const float rstd = rsqrtf(red[0] / (float)TK_ + 1e-5f); __syncthreads();
  if (t < TK_) att_out[(size_t)bq * TK_ + t] = (oav - omean) * rstd;

  for (int d = t; d < D_; d += 256) {                   // weight = softmax @ src_key
    float w = 0.f;
    for (int k = 0; k < TK_; ++k)
      w += sm[k] * src_key[(size_t)(b * TK_ + k) * D_ + d];
    wgt[d] = w * tgtm;
  }
  __syncthreads();
  float pa = 0.f;                                       // p = sigmoid(weight . W_lin + b)
  for (int d = t; d < D_; d += 256) pa += wgt[d] * W_lin[d];
  red[t] = pa; __syncthreads();
  for (int o = 128; o > 0; o >>= 1) { if (t < o) red[t] += red[t + o]; __syncthreads(); }
  if (t == 0) p_out[bq] = 1.f / (1.f + __expf(-(red[0] + b_lin[0])));
}

// ---------------------------------------------------------------- finalize + scatter
__global__ void k_finalize(float* __restrict__ out, const float* __restrict__ stats,
                           const float* __restrict__ p_ws) {
  const int bq = blockIdx.y;
  const int e  = blockIdx.x * 256 + threadIdx.x;
  if (e >= EXT_) return;
  const size_t idx = (size_t)bq * EXT_ + e;
  float v = 0.f;
  if (e < V_) v = p_ws[bq] * (out[idx] - stats[bq * 2]) * stats[bq * 2 + 1];
  out[idx] = v;
}

__global__ void k_scatter(float* __restrict__ out, const long long* __restrict__ map,
                          const float* __restrict__ att, const float* __restrict__ p_ws) {
  const int bq = blockIdx.x, k = threadIdx.x;
  const int b = bq >> 6;
  const long long e = map[b * TK_ + k];
  atomicAdd(&out[(size_t)bq * EXT_ + (size_t)e],
            (1.f - p_ws[bq]) * att[(size_t)bq * TK_ + k]);
}

// ---------------------------------------------------------------- launcher
extern "C" void kernel_launch(void* const* d_in, const int* in_sizes, int n_in,
                              void* d_out, int out_size, void* d_ws, size_t ws_size,
                              hipStream_t stream) {
  (void)in_sizes; (void)n_in; (void)out_size; (void)ws_size;
  const float*     tgt    = (const float*)d_in[0];
  const float*     src    = (const float*)d_in[1];
  const long long* midx   = (const long long*)d_in[2];   // int64 in reference
  const float*     W_out  = (const float*)d_in[3];
  const float*     b_out  = (const float*)d_in[4];
  const float*     W_attn = (const float*)d_in[5];
  const float*     b_attn = (const float*)d_in[6];
  const float*     v_w    = (const float*)d_in[7];
  const float*     W_lin  = (const float*)d_in[8];
  const float*     b_lin  = (const float*)d_in[9];
  float* out = (float*)d_out;

  char* ws = (char*)d_ws;
  unsigned short* wsA  = (unsigned short*)(ws + 0);          //  512x512 bf16
  unsigned short* wsW  = (unsigned short*)(ws + 524288);     // 32000x512 bf16
  unsigned short* wsWa = (unsigned short*)(ws + 33292288);   //  512x1024 bf16
  unsigned short* wsS  = (unsigned short*)(ws + 34340864);   // 1024x512 bf16
  float* qproj = (float*)(ws + 35389440);                    //  512x512 f32
  float* kproj = (float*)(ws + 36438016);                    // 1024x512 f32
  float* stats = (float*)(ws + 38535168);                    //  512x2
  float* tmask = (float*)(ws + 38539264);                    //  512
  float* smask = (float*)(ws + 38541312);                    // 1024
  float* attw  = (float*)(ws + 38545408);                    //  512x128
  float* pw    = (float*)(ws + 38807552);                    //  512

  k_convert<<<256,  256, 0, stream>>>(tgt,    wsA,  MROWS * D_);
  k_convert<<<2048, 256, 0, stream>>>(W_out,  wsW,  V_ * D_);
  k_convert<<<256,  256, 0, stream>>>(W_attn, wsWa, D_ * 2 * D_);
  k_convert<<<256,  256, 0, stream>>>(src,    wsS,  KROWS * D_);
  k_masks<<<MROWS + KROWS, 256, 0, stream>>>(tgt, src, tmask, smask);

  // logits -> d_out (ldc = EXT so the padded layout is already in place)
  k_gemm_bf16<<<dim3(V_ / 256, MROWS / 64), 256, 0, stream>>>(
      wsA, D_, wsW, D_, out, EXT_, D_, b_out);
  // q_proj = tgt @ Wq^T   (Wq rows = W_attn rows, cols [0,512))
  k_gemm_bf16<<<dim3(D_ / 256, MROWS / 64), 256, 0, stream>>>(
      wsA, D_, wsWa, 2 * D_, qproj, D_, D_, nullptr);
  // k_proj = src @ Wk^T   (Wk rows = W_attn rows, cols [512,1024))
  k_gemm_bf16<<<dim3(D_ / 256, KROWS / 64), 256, 0, stream>>>(
      wsS, D_, wsWa + D_, 2 * D_, kproj, D_, D_, nullptr);

  k_rowstats<<<MROWS, 256, 0, stream>>>(out, stats);
  k_attention<<<MROWS, 256, 0, stream>>>(qproj, kproj, src, b_attn, v_w,
                                         W_lin, b_lin, tmask, smask, attw, pw);
  k_finalize<<<dim3((EXT_ + 255) / 256, MROWS), 256, 0, stream>>>(out, stats, pw);
  k_scatter<<<MROWS, TK_, 0, stream>>>(out, midx, attw, pw);
}
